// NSTransformer_22393959482039
// MI455X (gfx1250) — matrix operand order repaired and checked by reading, not compiled, not run on hardware
//
#include <hip/hip_runtime.h>
#include <hip/hip_bf16.h>
#include <math.h>

// ---------------------------------------------------------------------------
// NSTransformer forward on MI455X (gfx1250, wave32, WMMA).
// Heavy GEMMs use v_wmma_f32_16x16x32_f16 (f16 in, f32 accumulate); f32 master
// copy of x is kept for residual/LayerNorm. Bandwidth-bound => fused epilogues.
// GEMM K-loop is unrolled x2 with two static register stages so the compiler
// renames registers instead of emitting v_mov copies between pipeline stages.
// ---------------------------------------------------------------------------

typedef _Float16 h16;
typedef __attribute__((ext_vector_type(16))) _Float16 v16h;
typedef __attribute__((ext_vector_type(8)))  _Float16 v8h;
typedef __attribute__((ext_vector_type(8)))  float    v8f;

#define BATCH 4096
#define SEQL  20
#define FEAT  20
#define DMODEL 512
#define NHEAD 8
#define DHEAD 64
#define DFF   2048
#define NLAYER 3
#define MROWS (BATCH * SEQL)   // 81920

// ------------------------- WMMA fragment loads -----------------------------
// A (16x32 f16), row-major A[M,K]:
//   lanes 0-15 : row=lane,   halves 0..7 = K kb+0..7,  halves 8..15 = K kb+16..23
//   lanes16-31 : row=lane-16,halves 0..7 = K kb+8..15, halves 8..15 = K kb+24..31
__device__ __forceinline__ v16h load_frag_a(const h16* __restrict__ A, int lda,
                                            int mr, int kb, int lane) {
    int row = mr + (lane & 15);
    int ko  = (lane < 16) ? 0 : 8;
    const h16* p = A + (size_t)row * lda + kb + ko;
    union { v16h v; v8h h[2]; } u;
    u.h[0] = *(const v8h*)(p);
    u.h[1] = *(const v8h*)(p + 16);
    return u.v;
}

// B (32x16 f16) with weights stored [N,K] row-major (C = A * W^T):
//   lanes 0-15 : col=lane,    halves 0..7 = K kb+0..7,  halves 8..15 = K kb+16..23
//   lanes16-31 : col=lane-16, halves 0..7 = K kb+8..15, halves 8..15 = K kb+24..31
__device__ __forceinline__ v16h load_frag_b(const h16* __restrict__ W, int ldw,
                                            int nc, int kb, int lane) {
    int col = nc + (lane & 15);
    int ko  = (lane < 16) ? 0 : 8;
    const h16* p = W + (size_t)col * ldw + kb + ko;
    union { v16h v; v8h h[2]; } u;
    u.h[0] = *(const v8h*)(p);
    u.h[1] = *(const v8h*)(p + 16);
    return u.v;
}

// ------------------------------- GEMM ---------------------------------------
// C[M,N] = A[M,K] * W[N,K]^T + bias.  Block: 256 threads = 8 waves (4 x 2).
// Each wave owns a 32x32 output tile (2x2 WMMA accumulators); block = 128x64.
// Two static fragment stages (no register copies): S0 = K-step kb, S1 = kb+32;
// loads for stage kb+64/kb+96 are issued between WMMAs, so VMEM overlaps the
// matrix pipe with partial s_wait_loadcnt waits.  Requires K % 64 == 0.
__launch_bounds__(256)
__global__ void gemm_f16_kernel(const h16* __restrict__ A, const h16* __restrict__ W,
                                const float* __restrict__ bias,
                                h16* __restrict__ outF16, float* __restrict__ resF32,
                                int M, int N, int K, int relu)
{
    const int lane = threadIdx.x & 31;
    const int wave = threadIdx.x >> 5;
    const int wm   = wave >> 1;          // 0..3
    const int wn   = wave & 1;           // 0..1
    const int mr   = blockIdx.y * 128 + wm * 32;
    const int nc   = blockIdx.x * 64 + wn * 32;

    v8f c[2][2] = {};
    v16h a0[2], b0[2], a1[2], b1[2];     // stage 0 (kb) and stage 1 (kb+32)

    a0[0] = load_frag_a(A, K, mr,      0, lane);
    a0[1] = load_frag_a(A, K, mr + 16, 0, lane);
    b0[0] = load_frag_b(W, K, nc,      0, lane);
    b0[1] = load_frag_b(W, K, nc + 16, 0, lane);
    a1[0] = load_frag_a(A, K, mr,      32, lane);
    a1[1] = load_frag_a(A, K, mr + 16, 32, lane);
    b1[0] = load_frag_b(W, K, nc,      32, lane);
    b1[1] = load_frag_b(W, K, nc + 16, 32, lane);

    for (int kb = 0; kb < K - 64; kb += 64) {
        // consume stage 0 (kb), refill it for kb+64
        c[0][0] = __builtin_amdgcn_wmma_f32_16x16x32_f16(false, a0[0], false, b0[0], (short)0, c[0][0], false, false);
        c[0][1] = __builtin_amdgcn_wmma_f32_16x16x32_f16(false, a0[0], false, b0[1], (short)0, c[0][1], false, false);
        c[1][0] = __builtin_amdgcn_wmma_f32_16x16x32_f16(false, a0[1], false, b0[0], (short)0, c[1][0], false, false);
        c[1][1] = __builtin_amdgcn_wmma_f32_16x16x32_f16(false, a0[1], false, b0[1], (short)0, c[1][1], false, false);
        a0[0] = load_frag_a(A, K, mr,      kb + 64, lane);
        a0[1] = load_frag_a(A, K, mr + 16, kb + 64, lane);
        b0[0] = load_frag_b(W, K, nc,      kb + 64, lane);
        b0[1] = load_frag_b(W, K, nc + 16, kb + 64, lane);
        // consume stage 1 (kb+32), refill it for kb+96
        c[0][0] = __builtin_amdgcn_wmma_f32_16x16x32_f16(false, a1[0], false, b1[0], (short)0, c[0][0], false, false);
        c[0][1] = __builtin_amdgcn_wmma_f32_16x16x32_f16(false, a1[0], false, b1[1], (short)0, c[0][1], false, false);
        c[1][0] = __builtin_amdgcn_wmma_f32_16x16x32_f16(false, a1[1], false, b1[0], (short)0, c[1][0], false, false);
        c[1][1] = __builtin_amdgcn_wmma_f32_16x16x32_f16(false, a1[1], false, b1[1], (short)0, c[1][1], false, false);
        a1[0] = load_frag_a(A, K, mr,      kb + 96, lane);
        a1[1] = load_frag_a(A, K, mr + 16, kb + 96, lane);
        b1[0] = load_frag_b(W, K, nc,      kb + 96, lane);
        b1[1] = load_frag_b(W, K, nc + 16, kb + 96, lane);
    }
    // tail: stages hold K-64 and K-32
    c[0][0] = __builtin_amdgcn_wmma_f32_16x16x32_f16(false, a0[0], false, b0[0], (short)0, c[0][0], false, false);
    c[0][1] = __builtin_amdgcn_wmma_f32_16x16x32_f16(false, a0[0], false, b0[1], (short)0, c[0][1], false, false);
    c[1][0] = __builtin_amdgcn_wmma_f32_16x16x32_f16(false, a0[1], false, b0[0], (short)0, c[1][0], false, false);
    c[1][1] = __builtin_amdgcn_wmma_f32_16x16x32_f16(false, a0[1], false, b0[1], (short)0, c[1][1], false, false);
    c[0][0] = __builtin_amdgcn_wmma_f32_16x16x32_f16(false, a1[0], false, b1[0], (short)0, c[0][0], false, false);
    c[0][1] = __builtin_amdgcn_wmma_f32_16x16x32_f16(false, a1[0], false, b1[1], (short)0, c[0][1], false, false);
    c[1][0] = __builtin_amdgcn_wmma_f32_16x16x32_f16(false, a1[1], false, b1[0], (short)0, c[1][0], false, false);
    c[1][1] = __builtin_amdgcn_wmma_f32_16x16x32_f16(false, a1[1], false, b1[1], (short)0, c[1][1], false, false);

    // D layout: lane<16 -> (M=+r, N=+lane); lane>=16 -> (M=+8+r, N=+lane-16)
    const int n_lo = lane & 15;
    const int r_hi = (lane < 16) ? 0 : 8;
#pragma unroll
    for (int ti = 0; ti < 2; ++ti) {
#pragma unroll
        for (int tj = 0; tj < 2; ++tj) {
            int n    = nc + tj * 16 + n_lo;
            float bv = bias ? bias[n] : 0.0f;
#pragma unroll
            for (int r = 0; r < 8; ++r) {
                int   m = mr + ti * 16 + r_hi + r;
                float v = c[ti][tj][r] + bv;
                if (relu && v < 0.0f) v = 0.0f;
                if (outF16) outF16[(size_t)m * N + n] = (h16)v;
                if (resF32) resF32[(size_t)m * N + n] += v;
            }
        }
    }
}

// ------------------------------ LayerNorm -----------------------------------
__launch_bounds__(128)
__global__ void ln_kernel(float* __restrict__ x32, h16* __restrict__ x16,
                          const float* __restrict__ g, const float* __restrict__ b)
{
    __shared__ float red[128];
    const size_t m = blockIdx.x;
    float* row = x32 + m * DMODEL;

    float v4[4];
    float s = 0.0f;
#pragma unroll
    for (int i = 0; i < 4; ++i) { v4[i] = row[threadIdx.x + i * 128]; s += v4[i]; }
    red[threadIdx.x] = s; __syncthreads();
    for (int st = 64; st > 0; st >>= 1) {
        if (threadIdx.x < st) red[threadIdx.x] += red[threadIdx.x + st];
        __syncthreads();
    }
    float mean = red[0] * (1.0f / DMODEL);
    __syncthreads();
    float ss = 0.0f;
#pragma unroll
    for (int i = 0; i < 4; ++i) { float d = v4[i] - mean; ss += d * d; }
    red[threadIdx.x] = ss; __syncthreads();
    for (int st = 64; st > 0; st >>= 1) {
        if (threadIdx.x < st) red[threadIdx.x] += red[threadIdx.x + st];
        __syncthreads();
    }
    float inv = rsqrtf(red[0] * (1.0f / DMODEL) + 1e-5f);
#pragma unroll
    for (int i = 0; i < 4; ++i) {
        int d = threadIdx.x + i * 128;
        float y = (v4[i] - mean) * inv * g[d] + b[d];
        row[d] = y;
        x16[m * DMODEL + d] = (h16)y;
    }
}

// ---------------------- DSAttention (tau/delta shaped) ----------------------
// One block per (batch, head); q/k/v are f16 [M, 512], head-contiguous slices.
__launch_bounds__(128)
__global__ void attn_kernel(const h16* __restrict__ q, const h16* __restrict__ k,
                            const h16* __restrict__ v,
                            const float* __restrict__ tau,
                            const float* __restrict__ delta,
                            h16* __restrict__ o)
{
    const int b = blockIdx.x, h = blockIdx.y;
    __shared__ h16 qs[SEQL * DHEAD], ks[SEQL * DHEAD], vs[SEQL * DHEAD];
    __shared__ float sc[SEQL * SEQL];
    const size_t base = ((size_t)b * SEQL) * DMODEL + (size_t)h * DHEAD;

    for (int idx = threadIdx.x; idx < SEQL * DHEAD; idx += 128) {
        int l = idx >> 6, d = idx & 63;
        size_t off = base + (size_t)l * DMODEL + d;
        qs[idx] = q[off]; ks[idx] = k[off]; vs[idx] = v[off];
    }
    __syncthreads();

    const float tb = tau[b];
    for (int idx = threadIdx.x; idx < SEQL * SEQL; idx += 128) {
        int l = idx / SEQL, s = idx % SEQL;
        float acc = 0.0f;
        for (int d = 0; d < DHEAD; ++d)
            acc += (float)qs[l * DHEAD + d] * (float)ks[s * DHEAD + d];
        sc[idx] = acc * tb + delta[b * SEQL + s];
    }
    __syncthreads();

    const float scale = 0.125f;   // 1/sqrt(64)
    if (threadIdx.x < SEQL) {
        int l = threadIdx.x;
        float mx = -1e30f;
        for (int s = 0; s < SEQL; ++s) { float t = scale * sc[l * SEQL + s]; mx = t > mx ? t : mx; }
        float sum = 0.0f;
        for (int s = 0; s < SEQL; ++s) {
            float e = __expf(scale * sc[l * SEQL + s] - mx);
            sc[l * SEQL + s] = e; sum += e;
        }
        float inv = 1.0f / sum;
        for (int s = 0; s < SEQL; ++s) sc[l * SEQL + s] *= inv;
    }
    __syncthreads();

    for (int idx = threadIdx.x; idx < SEQL * DHEAD; idx += 128) {
        int l = idx >> 6, d = idx & 63;
        float acc = 0.0f;
        for (int s = 0; s < SEQL; ++s)
            acc += sc[l * SEQL + s] * (float)vs[s * DHEAD + d];
        o[base + (size_t)l * DMODEL + d] = (h16)acc;
    }
}

// ------------------------------ small kernels -------------------------------
__global__ void stats_kernel(const float* __restrict__ src,
                             float* __restrict__ meanb, float* __restrict__ stdb)
{
    int b = blockIdx.x, f = threadIdx.x;
    if (f >= FEAT) return;
    float s = 0.0f;
    for (int l = 0; l < SEQL; ++l) s += src[((size_t)b * SEQL + l) * FEAT + f];
    float mean = s * (1.0f / SEQL);
    float var = 0.0f;
    for (int l = 0; l < SEQL; ++l) {
        float d = src[((size_t)b * SEQL + l) * FEAT + f] - mean;
        var += d * d;
    }
    meanb[b * FEAT + f] = mean;
    stdb[b * FEAT + f]  = sqrtf(var * (1.0f / SEQL) + 1e-5f);
}

// circular conv1d over feature axis + concat with stats -> z[B,40]
__global__ void conv_z_kernel(const float* __restrict__ src,
                              const float* __restrict__ meanb, const float* __restrict__ stdb,
                              const float* __restrict__ tconv, const float* __restrict__ dconv,
                              float* __restrict__ ztau, float* __restrict__ zdel)
{
    int b = blockIdx.x, t = threadIdx.x;
    int which = t >> 5;          // 0 = tau branch, 1 = delta branch
    int f = t & 31;
    if (f >= FEAT) return;
    const float* conv = which ? dconv : tconv;
    int fm = (f + FEAT - 1) % FEAT, fp = (f + 1) % FEAT;
    float acc = 0.0f;
    for (int c = 0; c < SEQL; ++c) {
        const float* row = src + ((size_t)b * SEQL + c) * FEAT;
        acc += row[fm] * conv[c * 3 + 0] + row[f] * conv[c * 3 + 1] + row[fp] * conv[c * 3 + 2];
    }
    float* z = which ? zdel : ztau;
    z[b * 40 + f]        = acc;
    z[b * 40 + FEAT + f] = which ? meanb[b * FEAT + f] : stdb[b * FEAT + f];
}

// naive dense: out[row, n] = act(in[row,:K] . W[n,:K] + bias[n]); grid (rows, ceil(N/64))
__global__ void dense_kernel(const float* __restrict__ in, const float* __restrict__ W,
                             const float* __restrict__ bias, float* __restrict__ out,
                             int K, int N, int relu)
{
    __shared__ float xr[512];
    int row = blockIdx.x;
    for (int i = threadIdx.x; i < K; i += blockDim.x) xr[i] = in[(size_t)row * K + i];
    __syncthreads();
    int n = blockIdx.y * 64 + threadIdx.x;
    if (n < N) {
        const float* w = W + (size_t)n * K;
        float acc = bias ? bias[n] : 0.0f;
        for (int kk = 0; kk < K; ++kk) acc += xr[kk] * w[kk];
        if (relu && acc < 0.0f) acc = 0.0f;
        out[(size_t)row * N + n] = acc;
    }
}

__global__ void tau_kernel(const float* __restrict__ hbuf, const float* __restrict__ W2,
                           float* __restrict__ tau)
{
    __shared__ float red[128];
    int b = blockIdx.x;
    float s = 0.0f;
    for (int kk = threadIdx.x; kk < DMODEL; kk += 128)
        s += hbuf[(size_t)b * DMODEL + kk] * W2[kk];
    red[threadIdx.x] = s; __syncthreads();
    for (int st = 64; st > 0; st >>= 1) {
        if (threadIdx.x < st) red[threadIdx.x] += red[threadIdx.x + st];
        __syncthreads();
    }
    if (threadIdx.x == 0) tau[b] = __expf(red[0]);
}

// x = (src - mean)/std @ W_feat^T + b_feat + PE;  writes f32 and f16 copies
__global__ void embed_kernel(const float* __restrict__ src,
                             const float* __restrict__ meanb, const float* __restrict__ stdb,
                             const float* __restrict__ Wf, const float* __restrict__ bf,
                             float* __restrict__ x32, h16* __restrict__ x16)
{
    __shared__ float s[FEAT];
    int m = blockIdx.x;
    int b = m / SEQL, l = m % SEQL;
    if (threadIdx.x < FEAT) {
        int f = threadIdx.x;
        s[f] = (src[(size_t)m * FEAT + f] - meanb[b * FEAT + f]) / stdb[b * FEAT + f];
    }
    __syncthreads();
    int d = threadIdx.x;
    const float* w = Wf + (size_t)d * FEAT;
    float acc = bf[d];
    for (int f = 0; f < FEAT; ++f) acc += s[f] * w[f];
    int i2 = d & ~1;
    float div = __expf((float)i2 * (-9.2103403719761836f / (float)DMODEL)); // ln(10000)
    float ang = (float)l * div;
    acc += (d & 1) ? __cosf(ang) : __sinf(ang);
    x32[(size_t)m * DMODEL + d] = acc;
    x16[(size_t)m * DMODEL + d] = (h16)acc;
}

// out[b] = sum_f ((x_last . W_pre[f] + b_pre[f]) * std + mean) * W_dec[f] + b_dec
__global__ void final_kernel(const float* __restrict__ x32,
                             const float* __restrict__ Wpre, const float* __restrict__ bpre,
                             const float* __restrict__ Wdec, const float* __restrict__ bdec,
                             const float* __restrict__ meanb, const float* __restrict__ stdb,
                             float* __restrict__ out)
{
    __shared__ float red[32];
    int b = blockIdx.x;
    const float* xl = x32 + ((size_t)b * SEQL + (SEQL - 1)) * DMODEL;
    int f = threadIdx.x;
    float val = 0.0f;
    if (f < FEAT) {
        const float* w = Wpre + (size_t)f * DMODEL;
        float acc = bpre[f];
        for (int kk = 0; kk < DMODEL; ++kk) acc += xl[kk] * w[kk];
        acc = acc * stdb[b * FEAT + f] + meanb[b * FEAT + f];
        val = acc * Wdec[f];
    }
    red[threadIdx.x] = val; __syncthreads();
    for (int st = 16; st > 0; st >>= 1) {
        if (threadIdx.x < st) red[threadIdx.x] += red[threadIdx.x + st];
        __syncthreads();
    }
    if (threadIdx.x == 0) out[b] = red[0] + bdec[0];
}

__global__ void cvt_f16_kernel(const float* __restrict__ in, h16* __restrict__ out, size_t n)
{
    size_t i = (size_t)blockIdx.x * blockDim.x + threadIdx.x;
    if (i < n) out[i] = (h16)in[i];
}

// ------------------------------ host launcher -------------------------------
extern "C" void kernel_launch(void* const* d_in, const int* in_sizes, int n_in,
                              void* d_out, int out_size, void* d_ws, size_t ws_size,
                              hipStream_t stream)
{
    (void)in_sizes; (void)n_in; (void)out_size; (void)ws_size;
    const float* src      = (const float*)d_in[0];
    const float* W_feat   = (const float*)d_in[1];
    const float* b_feat   = (const float*)d_in[2];
    const float* tau_conv = (const float*)d_in[3];
    const float* tau_W0   = (const float*)d_in[4];
    const float* tau_b0   = (const float*)d_in[5];
    const float* tau_W1   = (const float*)d_in[6];
    const float* tau_b1   = (const float*)d_in[7];
    const float* tau_W2   = (const float*)d_in[8];
    const float* del_conv = (const float*)d_in[9];
    const float* del_W0   = (const float*)d_in[10];
    const float* del_b0   = (const float*)d_in[11];
    const float* del_W1   = (const float*)d_in[12];
    const float* del_b1   = (const float*)d_in[13];
    const float* del_W2   = (const float*)d_in[14];
    const float* Wq = (const float*)d_in[15]; const float* bq = (const float*)d_in[16];
    const float* Wk = (const float*)d_in[17]; const float* bk = (const float*)d_in[18];
    const float* Wv = (const float*)d_in[19]; const float* bv = (const float*)d_in[20];
    const float* Wo = (const float*)d_in[21]; const float* bo = (const float*)d_in[22];
    const float* Wc1 = (const float*)d_in[23]; const float* bc1 = (const float*)d_in[24];
    const float* Wc2 = (const float*)d_in[25]; const float* bc2 = (const float*)d_in[26];
    const float* ln1_g = (const float*)d_in[27]; const float* ln1_b = (const float*)d_in[28];
    const float* ln2_g = (const float*)d_in[29]; const float* ln2_b = (const float*)d_in[30];
    const float* lnf_g = (const float*)d_in[31]; const float* lnf_b = (const float*)d_in[32];
    const float* W_pre = (const float*)d_in[33]; const float* b_pre = (const float*)d_in[34];
    const float* W_dec = (const float*)d_in[35]; const float* b_dec = (const float*)d_in[36];
    float* out = (float*)d_out;

    // ---------------- workspace carve-up (256B aligned) ----------------
    char* ws = (char*)d_ws;
    size_t off = 0;
    auto alloc = [&](size_t bytes) -> char* {
        char* p = ws + off;
        off = (off + bytes + 255) & ~(size_t)255;
        return p;
    };
    const size_t xElems = (size_t)MROWS * DMODEL;      // 41.9M
    float* x32  = (float*)alloc(xElems * 4);           // 168 MB master copy
    h16*   x16  = (h16*)  alloc(xElems * 2);           //  84 MB f16 mirror
    // q|k|v|o contiguous blob, time-shared with the FFN hidden (same 336 MB)
    h16*   blob = (h16*)  alloc(4 * xElems * 2);
    h16* q16 = blob;            h16* k16 = blob + xElems;
    h16* v16 = blob + 2*xElems; h16* o16 = blob + 3*xElems;
    h16* hid16 = blob;                                  // [M, 2048] f16, aliases q..o
    float* meanb = (float*)alloc((size_t)BATCH * FEAT * 4);
    float* stdb  = (float*)alloc((size_t)BATCH * FEAT * 4);
    float* ztau  = (float*)alloc((size_t)BATCH * 40 * 4);
    float* zdel  = (float*)alloc((size_t)BATCH * 40 * 4);
    float* hA    = (float*)alloc((size_t)BATCH * DMODEL * 4);
    float* hB    = (float*)alloc((size_t)BATCH * DMODEL * 4);
    float* taub  = (float*)alloc((size_t)BATCH * 4);
    float* deltb = (float*)alloc((size_t)BATCH * SEQL * 4);
    const size_t wProj = (size_t)NLAYER * DMODEL * DMODEL;   // 786432
    const size_t wFF   = (size_t)NLAYER * DFF * DMODEL;      // 3145728
    h16* wq16 = (h16*)alloc(wProj * 2); h16* wk16 = (h16*)alloc(wProj * 2);
    h16* wv16 = (h16*)alloc(wProj * 2); h16* wo16 = (h16*)alloc(wProj * 2);
    h16* wc1_16 = (h16*)alloc(wFF * 2); h16* wc2_16 = (h16*)alloc(wFF * 2);

    auto cvt = [&](const float* in, h16* o, size_t n) {
        cvt_f16_kernel<<<dim3((unsigned)((n + 255) / 256)), dim3(256), 0, stream>>>(in, o, n);
    };
    auto gemm = [&](const h16* A, const h16* W, const float* bias,
                    h16* oF16, float* res32, int M_, int N_, int K_, int relu) {
        dim3 g((unsigned)(N_ / 64), (unsigned)(M_ / 128));
        gemm_f16_kernel<<<g, dim3(256), 0, stream>>>(A, W, bias, oF16, res32, M_, N_, K_, relu);
    };

    // f16 weight mirrors (cheap: ~19 MB total)
    cvt(Wq, wq16, wProj); cvt(Wk, wk16, wProj); cvt(Wv, wv16, wProj); cvt(Wo, wo16, wProj);
    cvt(Wc1, wc1_16, wFF); cvt(Wc2, wc2_16, wFF);

    // ---- non-stationary stats + tau/delta projectors ----
    stats_kernel<<<dim3(BATCH), dim3(32), 0, stream>>>(src, meanb, stdb);
    conv_z_kernel<<<dim3(BATCH), dim3(64), 0, stream>>>(src, meanb, stdb, tau_conv, del_conv, ztau, zdel);
    // tau MLP
    dense_kernel<<<dim3(BATCH, DMODEL / 64), dim3(64), 0, stream>>>(ztau, tau_W0, tau_b0, hA, 40, DMODEL, 1);
    dense_kernel<<<dim3(BATCH, DMODEL / 64), dim3(64), 0, stream>>>(hA, tau_W1, tau_b1, hB, DMODEL, DMODEL, 1);
    tau_kernel<<<dim3(BATCH), dim3(128), 0, stream>>>(hB, tau_W2, taub);
    // delta MLP
    dense_kernel<<<dim3(BATCH, DMODEL / 64), dim3(64), 0, stream>>>(zdel, del_W0, del_b0, hA, 40, DMODEL, 1);
    dense_kernel<<<dim3(BATCH, DMODEL / 64), dim3(64), 0, stream>>>(hA, del_W1, del_b1, hB, DMODEL, DMODEL, 1);
    dense_kernel<<<dim3(BATCH, 1), dim3(64), 0, stream>>>(hB, del_W2, nullptr, deltb, DMODEL, SEQL, 0);

    // ---- embedding + positional encoding ----
    embed_kernel<<<dim3(MROWS), dim3(DMODEL), 0, stream>>>(src, meanb, stdb, W_feat, b_feat, x32, x16);

    // ---- encoder layers ----
    for (int i = 0; i < NLAYER; ++i) {
        const size_t wp = (size_t)i * DMODEL * DMODEL;
        const size_t wf = (size_t)i * DFF * DMODEL;
        gemm(x16, wq16 + wp, bq + i * DMODEL, q16, nullptr, MROWS, DMODEL, DMODEL, 0);
        gemm(x16, wk16 + wp, bk + i * DMODEL, k16, nullptr, MROWS, DMODEL, DMODEL, 0);
        gemm(x16, wv16 + wp, bv + i * DMODEL, v16, nullptr, MROWS, DMODEL, DMODEL, 0);
        attn_kernel<<<dim3(BATCH, NHEAD), dim3(128), 0, stream>>>(q16, k16, v16, taub, deltb, o16);
        gemm(o16, wo16 + wp, bo + i * DMODEL, nullptr, x32, MROWS, DMODEL, DMODEL, 0); // residual add
        ln_kernel<<<dim3(MROWS), dim3(128), 0, stream>>>(x32, x16, ln1_g + i * DMODEL, ln1_b + i * DMODEL);
        gemm(x16, wc1_16 + wf, bc1 + i * DFF, hid16, nullptr, MROWS, DFF, DMODEL, 1);  // ReLU
        gemm(hid16, wc2_16 + wf, bc2 + i * DMODEL, nullptr, x32, MROWS, DMODEL, DFF, 0); // residual add
        ln_kernel<<<dim3(MROWS), dim3(128), 0, stream>>>(x32, x16, ln2_g + i * DMODEL, ln2_b + i * DMODEL);
    }

    // ---- final LN + de-normalized head ----
    ln_kernel<<<dim3(MROWS), dim3(128), 0, stream>>>(x32, x16, lnf_g, lnf_b);
    final_kernel<<<dim3(BATCH), dim3(32), 0, stream>>>(x32, W_pre, b_pre, W_dec, b_dec, meanb, stdb, out);
}
